// MoELayer_12850542149814
// MI455X (gfx1250) — matrix operand order repaired and checked by reading, not compiled, not run on hardware
//
#include <hip/hip_runtime.h>
#include <hip/hip_bf16.h>
#include <math.h>

// MoE: E=8 experts, H=1024, I=2048, top-2; T = 4*2048 = 8192 tokens.
#define E_      8
#define H_      1024
#define I_      2048
#define TOKENS  8192
#define NA      (2 * TOKENS)   // total (token, expert) assignments
#define KPAD    40             // bf16 elems per LDS row: 80B, multiple of 16B

typedef __bf16 bf16_t;
typedef __bf16 v16bf __attribute__((ext_vector_type(16)));
typedef float  v8f   __attribute__((ext_vector_type(8)));
typedef unsigned int u32x4 __attribute__((ext_vector_type(4)));
typedef int          i32x8 __attribute__((ext_vector_type(8)));
typedef int          i32x4 __attribute__((ext_vector_type(4)));

#if defined(__has_builtin)
#if __has_builtin(__builtin_amdgcn_tensor_load_to_lds) && __has_builtin(__builtin_amdgcn_s_wait_tensorcnt)
#define USE_TDM 1
#endif
#endif
#ifndef USE_TDM
#define USE_TDM 0
#endif

#if USE_TDM
// TDM 2D tile load: [32 rows(K) x 128 lines] of bf16 from a row-major tensor
// whose lines are row_len elems long, into LDS with an 80B line pitch
// (64B data + 16B pad via D# pad fields). Issued by one wave; TENSORcnt-tracked.
__device__ __forceinline__ void tdm_load_2d(unsigned lds_byte_addr,
                                            const bf16_t* gptr, int row_len) {
  unsigned long long ga = (unsigned long long)(uintptr_t)gptr;
  u32x4 g0;
  g0[0] = 1u;                                         // count=1, load, no gather
  g0[1] = lds_byte_addr;                              // lds_addr
  g0[2] = (unsigned)ga;                               // global_addr[31:0]
  g0[3] = (unsigned)((ga >> 32) & 0x1FFFFFFu) | 0x80000000u; // addr[56:32]|type=2
  i32x8 g1;
  g1[0] = (1 << 16)   // data_size = 2 bytes
        | (1 << 20)   // pad_enable
        | (3 << 22)   // pad_interval: every 16 DWORDs (64B)
        | (3 << 25);  // pad_amount: 4 DWORDs (16B) -> 80B LDS pitch
  g1[1] = (row_len & 0xFFFF) << 16;                   // tensor_dim0[15:0]
  g1[2] = ((row_len >> 16) & 0xFFFF);                 // tensor_dim0[31:16], dim1 lo=0
  g1[3] = 0x10 | (32 << 16);                          // tensor_dim1=1<<20, tile_dim0=32
  g1[4] = 128;                                        // tile_dim1=128, tile_dim2=0
  g1[5] = row_len;                                    // tensor_dim0_stride[31:0]
  g1[6] = 0;
  g1[7] = 0;
  i32x4 z4 = {0, 0, 0, 0};
  i32x8 z8 = {0, 0, 0, 0, 0, 0, 0, 0};
  __builtin_amdgcn_tensor_load_to_lds(g0, g1, z4, z4, z8, 0);
}
#endif

// A-fragment (16x32 bf16): lanes 0-15 row M=lane, K 0-15;
// lanes 16-31 same row, K 16-31. Contiguous 16B chunks -> ds_load_b128.
__device__ __forceinline__ v16bf load_a_frag(const bf16_t* strip, int lane) {
  const int r = lane & 15, half = lane >> 4;
  const bf16_t* p = strip + r * KPAD;
  v16bf a;
#pragma unroll
  for (int j = 0; j < 8; ++j) {
    a[j]     = p[16 * half + j];
    a[8 + j] = p[16 * half + 8 + j];
  }
  return a;
}

// B-fragment (32x16 bf16): lane = column; lanes 0-15 hold K=0..15, lanes 16-31
// K=16..31. Tile sits K-contiguous per line in LDS -> 32 contiguous bytes/lane.
__device__ __forceinline__ v16bf load_b_frag(const bf16_t* colmaj, int lane) {
  const int n = lane & 15, half = lane >> 4;
  const bf16_t* p = colmaj + n * KPAD + 16 * half;
  v16bf b;
#pragma unroll
  for (int j = 0; j < 16; ++j) b[j] = p[j];
  return b;
}

// ---------------- bf16 pre-conversion ----------------
__global__ void convert_x_kernel(const float* __restrict__ src,
                                 bf16_t* __restrict__ dst, long n) {
  long i = ((long)blockIdx.x * blockDim.x + threadIdx.x) * 4;
  long stride = (long)gridDim.x * blockDim.x * 4;
  for (long j = i; j < n; j += stride) {
    float4 v = *(const float4*)(src + j);
    dst[j + 0] = (bf16_t)v.x; dst[j + 1] = (bf16_t)v.y;
    dst[j + 2] = (bf16_t)v.z; dst[j + 3] = (bf16_t)v.w;
  }
}

// src: [E][R][C] fp32 -> dst: [E][C][R] bf16 (per-expert transpose + convert)
__global__ void transpose_convert_kernel(const float* __restrict__ src,
                                         bf16_t* __restrict__ dst, int R, int C) {
  __shared__ float t[32][33];
  const long base = (long)blockIdx.z * R * C;
  const int c0 = blockIdx.x * 32, r0 = blockIdx.y * 32;
  const int tx = threadIdx.x & 31, ty = threadIdx.x >> 5;   // 256 thr: ty 0..7
#pragma unroll
  for (int j = 0; j < 32; j += 8)
    t[ty + j][tx] = src[base + (long)(r0 + ty + j) * C + c0 + tx];
  __syncthreads();
#pragma unroll
  for (int j = 0; j < 32; j += 8)
    dst[base + (long)(c0 + ty + j) * R + r0 + tx] = (bf16_t)t[tx][ty + j];
}

// ---------------- router: logits, top-2, softmax ----------------
__global__ void router_kernel(const float* __restrict__ x,
                              const float* __restrict__ Wr,
                              const float* __restrict__ br,
                              int* __restrict__ top_i, float* __restrict__ top_w) {
  int t = blockIdx.x * blockDim.x + threadIdx.x;
  if (t >= TOKENS) return;
  float acc[E_];
#pragma unroll
  for (int e = 0; e < E_; ++e) acc[e] = br[e];
  const float* xr = x + (long)t * H_;
  for (int h = 0; h < H_; ++h) {
    float xv = xr[h];
    const float* wr = Wr + (long)h * E_;
#pragma unroll
    for (int e = 0; e < E_; ++e) acc[e] = fmaf(xv, wr[e], acc[e]);
  }
  int i1 = 0; float v1 = acc[0];
#pragma unroll
  for (int e = 1; e < E_; ++e) if (acc[e] > v1) { v1 = acc[e]; i1 = e; }
  int i2 = -1; float v2 = -INFINITY;
#pragma unroll
  for (int e = 0; e < E_; ++e) if (e != i1 && acc[e] > v2) { v2 = acc[e]; i2 = e; }
  float q  = __expf(v2 - v1);      // v2 <= v1, stable
  float p1 = 1.0f / (1.0f + q);
  top_i[2 * t]     = i1; top_w[2 * t]     = p1;
  top_i[2 * t + 1] = i2; top_w[2 * t + 1] = 1.0f - p1;
}

__global__ void count_kernel(const int* __restrict__ top_i,
                             int* __restrict__ counts, int* __restrict__ offsets,
                             int* __restrict__ cursor) {
  __shared__ int c[E_];
  int tid = threadIdx.x;
  if (tid < E_) c[tid] = 0;
  __syncthreads();
  for (int a = tid; a < NA; a += blockDim.x) atomicAdd(&c[top_i[a]], 1);
  __syncthreads();
  if (tid == 0) {
    int s = 0;
    for (int e = 0; e < E_; ++e) { offsets[e] = s; s += c[e]; }
    offsets[E_] = s;
  }
  if (tid < E_) { counts[tid] = c[tid]; cursor[tid] = 0; }
}

__global__ void scatter_kernel(const int* __restrict__ top_i,
                               const float* __restrict__ top_w,
                               const int* __restrict__ offsets, int* __restrict__ cursor,
                               int* __restrict__ tokId, float* __restrict__ tokW) {
  int a = blockIdx.x * blockDim.x + threadIdx.x;
  if (a >= NA) return;
  int e = top_i[a];
  int slot = atomicAdd(&cursor[e], 1);
  int r = offsets[e] + slot;
  tokId[r] = a >> 1;
  tokW[r]  = top_w[a];
}

// ---------------- gate/up GEMMs + SiLU (double-buffered) ----------------
__global__ __launch_bounds__(256)
void gateup_kernel(const bf16_t* __restrict__ xb,
                   const bf16_t* __restrict__ Wgt,   // [E][I][H] bf16
                   const bf16_t* __restrict__ Wut,   // [E][I][H] bf16
                   const float* __restrict__ bg, const float* __restrict__ bu,
                   const int* __restrict__ offsets, const int* __restrict__ tokId,
                   bf16_t* __restrict__ hbuf) {
  const int e    = blockIdx.z;
  const int base = offsets[e];
  const int rows = offsets[e + 1] - base;
  const int r0   = blockIdx.y * 128;
  if (r0 >= rows) return;
  const int rowsValid = min(128, rows - r0);
  const int icol0 = blockIdx.x * 128;

  __shared__ __align__(16) bf16_t ldsX[2][128 * KPAD];
  __shared__ __align__(16) bf16_t ldsG[2][128 * KPAD];
  __shared__ __align__(16) bf16_t ldsU[2][128 * KPAD];
  __shared__ int toks[128];

  const int tid  = threadIdx.x;
  const int lane = tid & 31;
  const int wid  = tid >> 5;

  for (int r = tid; r < 128; r += 256)
    toks[r] = tokId[min(base + r0 + r, NA - 1)];
  __syncthreads();

  // fixed per-thread staging slots: rows sr0/sr1, 16B chunk at sc elems
  const int sr0 = tid >> 2, sr1 = sr0 + 64, sc = (tid & 3) * 8;
  const bf16_t* px0 = xb + (long)toks[sr0] * H_ + sc;
  const bf16_t* px1 = xb + (long)toks[sr1] * H_ + sc;
  const unsigned dx0 = sr0 * KPAD + sc, dx1 = sr1 * KPAD + sc;

  const bf16_t* wg_tile = Wgt + (long)e * I_ * H_ + (long)icol0 * H_;
  const bf16_t* wu_tile = Wut + (long)e * I_ * H_ + (long)icol0 * H_;
#if USE_TDM
  const unsigned aG[2] = {(unsigned)(uintptr_t)&ldsG[0][0], (unsigned)(uintptr_t)&ldsG[1][0]};
  const unsigned aU[2] = {(unsigned)(uintptr_t)&ldsU[0][0], (unsigned)(uintptr_t)&ldsU[1][0]};
#else
  const bf16_t* pg0 = wg_tile + (long)sr0 * H_ + sc;
  const bf16_t* pg1 = wg_tile + (long)sr1 * H_ + sc;
  const bf16_t* pu0 = wu_tile + (long)sr0 * H_ + sc;
  const bf16_t* pu1 = wu_tile + (long)sr1 * H_ + sc;
#endif

  v8f cg[8] = {}, cu[8] = {};

  // prologue: stage chunk k0=0 into buffer 0
#if USE_TDM
  if (wid == 0) { tdm_load_2d(aG[0], wg_tile, H_); tdm_load_2d(aU[0], wu_tile, H_); }
#else
  *(uint4*)&ldsG[0][dx0] = *(const uint4*)pg0;
  *(uint4*)&ldsG[0][dx1] = *(const uint4*)pg1;
  *(uint4*)&ldsU[0][dx0] = *(const uint4*)pu0;
  *(uint4*)&ldsU[0][dx1] = *(const uint4*)pu1;
#endif
  *(uint4*)&ldsX[0][dx0] = *(const uint4*)px0;
  *(uint4*)&ldsX[0][dx1] = *(const uint4*)px1;

  int cur = 0;
  for (int k0 = 0; k0 < H_; k0 += 32) {
    const int nxt = cur ^ 1;
    const bool hasNext = (k0 + 32) < H_;
#if USE_TDM
    __builtin_amdgcn_s_wait_tensorcnt(0);   // buffer `cur` tiles landed (wave0)
#endif
    __syncthreads();                        // everyone sees buffer `cur`

    uint4 a0 = {}, a1 = {};
#if !USE_TDM
    uint4 g0v = {}, g1v = {}, u0v = {}, u1v = {};
#endif
    if (hasNext) {
      const int kn = k0 + 32;
#if USE_TDM
      if (wid == 0) {
        tdm_load_2d(aG[nxt], wg_tile + kn, H_);
        tdm_load_2d(aU[nxt], wu_tile + kn, H_);
      }
#else
      g0v = *(const uint4*)(pg0 + kn); g1v = *(const uint4*)(pg1 + kn);
      u0v = *(const uint4*)(pu0 + kn); u1v = *(const uint4*)(pu1 + kn);
#endif
      a0 = *(const uint4*)(px0 + kn); a1 = *(const uint4*)(px1 + kn);
    }

    v16bf a = load_a_frag(&ldsX[cur][0] + wid * 16 * KPAD, lane);
#pragma unroll
    for (int n = 0; n < 8; ++n) {
      v16bf bgf  = load_b_frag(&ldsG[cur][0] + n * 16 * KPAD, lane);
      v16bf buf_ = load_b_frag(&ldsU[cur][0] + n * 16 * KPAD, lane);
      cg[n] = __builtin_amdgcn_wmma_f32_16x16x32_bf16(false, a, false, bgf,
                                                      (short)0, cg[n], false, false);
      cu[n] = __builtin_amdgcn_wmma_f32_16x16x32_bf16(false, a, false, buf_,
                                                      (short)0, cu[n], false, false);
    }

    if (hasNext) {
#if !USE_TDM
      *(uint4*)&ldsG[nxt][dx0] = g0v; *(uint4*)&ldsG[nxt][dx1] = g1v;
      *(uint4*)&ldsU[nxt][dx0] = u0v; *(uint4*)&ldsU[nxt][dx1] = u1v;
#endif
      *(uint4*)&ldsX[nxt][dx0] = a0;  *(uint4*)&ldsX[nxt][dx1] = a1;
    }
    cur = nxt;
  }

  const int half = lane >> 4, cn = lane & 15;
#pragma unroll
  for (int n = 0; n < 8; ++n) {
    int col = icol0 + n * 16 + cn;
    float bgv = bg[e * I_ + col];
    float buv = bu[e * I_ + col];
#pragma unroll
    for (int i = 0; i < 8; ++i) {
      int rl = wid * 16 + i + 8 * half;
      if (rl < rowsValid) {
        float g = cg[n][i] + bgv;
        float u = cu[n][i] + buv;
        float hv = (g / (1.0f + __expf(-g))) * u;   // silu(g) * u
        hbuf[(long)(base + r0 + rl) * I_ + col] = (bf16_t)hv;
      }
    }
  }
}

// ---------------- down GEMM, weighted scatter-add (double-buffered) ----------------
__global__ __launch_bounds__(256)
void down_kernel(const bf16_t* __restrict__ hbuf,
                 const bf16_t* __restrict__ Wdt,    // [E][H][I] bf16
                 const float* __restrict__ bd,
                 const int* __restrict__ offsets,
                 const int* __restrict__ tokId, const float* __restrict__ tokW,
                 float* __restrict__ out) {
  const int e    = blockIdx.z;
  const int base = offsets[e];
  const int rows = offsets[e + 1] - base;
  const int r0   = blockIdx.y * 128;
  if (r0 >= rows) return;
  const int rowsValid = min(128, rows - r0);
  const int hcol0 = blockIdx.x * 128;

  __shared__ __align__(16) bf16_t ldsA[2][128 * KPAD];
  __shared__ __align__(16) bf16_t ldsW[2][128 * KPAD];
  __shared__ int   toks[128];
  __shared__ float tws[128];

  const int tid  = threadIdx.x;
  const int lane = tid & 31;
  const int wid  = tid >> 5;

  for (int r = tid; r < 128; r += 256) {
    int g = min(base + r0 + r, NA - 1);
    toks[r] = tokId[g];
    tws[r]  = tokW[g];
  }
  __syncthreads();

  const int sr0 = tid >> 2, sr1 = sr0 + 64, sc = (tid & 3) * 8;
  const bf16_t* ph0 = hbuf + (long)min(base + r0 + sr0, NA - 1) * I_ + sc;
  const bf16_t* ph1 = hbuf + (long)min(base + r0 + sr1, NA - 1) * I_ + sc;
  const unsigned dx0 = sr0 * KPAD + sc, dx1 = sr1 * KPAD + sc;

  const bf16_t* wd_tile = Wdt + (long)e * H_ * I_ + (long)hcol0 * I_;
#if USE_TDM
  const unsigned aW[2] = {(unsigned)(uintptr_t)&ldsW[0][0], (unsigned)(uintptr_t)&ldsW[1][0]};
#else
  const bf16_t* pw0 = wd_tile + (long)sr0 * I_ + sc;
  const bf16_t* pw1 = wd_tile + (long)sr1 * I_ + sc;
#endif

  v8f c[8] = {};

  // prologue: stage chunk k0=0 into buffer 0
#if USE_TDM
  if (wid == 0) tdm_load_2d(aW[0], wd_tile, I_);
#else
  *(uint4*)&ldsW[0][dx0] = *(const uint4*)pw0;
  *(uint4*)&ldsW[0][dx1] = *(const uint4*)pw1;
#endif
  *(uint4*)&ldsA[0][dx0] = *(const uint4*)ph0;
  *(uint4*)&ldsA[0][dx1] = *(const uint4*)ph1;

  int cur = 0;
  for (int k0 = 0; k0 < I_; k0 += 32) {
    const int nxt = cur ^ 1;
    const bool hasNext = (k0 + 32) < I_;
#if USE_TDM
    __builtin_amdgcn_s_wait_tensorcnt(0);
#endif
    __syncthreads();

    uint4 a0 = {}, a1 = {};
#if !USE_TDM
    uint4 w0v = {}, w1v = {};
#endif
    if (hasNext) {
      const int kn = k0 + 32;
#if USE_TDM
      if (wid == 0) tdm_load_2d(aW[nxt], wd_tile + kn, I_);
#else
      w0v = *(const uint4*)(pw0 + kn); w1v = *(const uint4*)(pw1 + kn);
#endif
      a0 = *(const uint4*)(ph0 + kn); a1 = *(const uint4*)(ph1 + kn);
    }

    v16bf a = load_a_frag(&ldsA[cur][0] + wid * 16 * KPAD, lane);
#pragma unroll
    for (int n = 0; n < 8; ++n) {
      v16bf b = load_b_frag(&ldsW[cur][0] + n * 16 * KPAD, lane);
      c[n] = __builtin_amdgcn_wmma_f32_16x16x32_bf16(false, a, false, b,
                                                     (short)0, c[n], false, false);
    }

    if (hasNext) {
#if !USE_TDM
      *(uint4*)&ldsW[nxt][dx0] = w0v; *(uint4*)&ldsW[nxt][dx1] = w1v;
#endif
      *(uint4*)&ldsA[nxt][dx0] = a0;  *(uint4*)&ldsA[nxt][dx1] = a1;
    }
    cur = nxt;
  }

  const int half = lane >> 4, cn = lane & 15;
#pragma unroll
  for (int n = 0; n < 8; ++n) {
    int col = hcol0 + n * 16 + cn;
    float bdv = bd[e * H_ + col];
#pragma unroll
    for (int i = 0; i < 8; ++i) {
      int rl = wid * 16 + i + 8 * half;
      if (rl < rowsValid) {
        float y = (c[n][i] + bdv) * tws[rl];
        unsafeAtomicAdd(&out[(long)toks[rl] * H_ + col], y);
      }
    }
  }
}

// ---------------- launch ----------------
extern "C" void kernel_launch(void* const* d_in, const int* in_sizes, int n_in,
                              void* d_out, int out_size, void* d_ws, size_t ws_size,
                              hipStream_t stream) {
  const float* x  = (const float*)d_in[0];
  const float* Wg = (const float*)d_in[1];
  const float* bg = (const float*)d_in[2];
  const float* Wu = (const float*)d_in[3];
  const float* bu = (const float*)d_in[4];
  const float* Wd = (const float*)d_in[5];
  const float* bd = (const float*)d_in[6];
  const float* Wr = (const float*)d_in[7];
  const float* br = (const float*)d_in[8];
  float* out = (float*)d_out;

  const size_t MB = 1u << 20;
  char* w = (char*)d_ws;
  int*    counts  = (int*)(w + 0);
  int*    offsets = (int*)(w + 64);
  int*    cursor  = (int*)(w + 128);
  int*    top_i   = (int*)(w + 256);
  float*  top_w   = (float*)(w + 256 + 4 * (size_t)NA);
  int*    tokId   = (int*)(w + 256 + 8 * (size_t)NA);
  float*  tokW    = (float*)(w + 256 + 12 * (size_t)NA);
  bf16_t* hbuf    = (bf16_t*)(w + 1 * MB);     // 64 MB
  bf16_t* xb      = (bf16_t*)(w + 65 * MB);    // 16 MB
  bf16_t* Wgt     = (bf16_t*)(w + 81 * MB);    // 32 MB
  bf16_t* Wut     = (bf16_t*)(w + 113 * MB);   // 32 MB
  bf16_t* Wdt     = (bf16_t*)(w + 145 * MB);   // 32 MB  (total 177 MB)

  (void)hipMemsetAsync(d_out, 0, (size_t)out_size * sizeof(float), stream);

  convert_x_kernel<<<2048, 256, 0, stream>>>(x, xb, (long)TOKENS * H_);
  transpose_convert_kernel<<<dim3(I_ / 32, H_ / 32, E_), 256, 0, stream>>>(Wg, Wgt, H_, I_);
  transpose_convert_kernel<<<dim3(I_ / 32, H_ / 32, E_), 256, 0, stream>>>(Wu, Wut, H_, I_);
  transpose_convert_kernel<<<dim3(H_ / 32, I_ / 32, E_), 256, 0, stream>>>(Wd, Wdt, I_, H_);

  router_kernel<<<(TOKENS + 127) / 128, 128, 0, stream>>>(x, Wr, br, top_i, top_w);
  count_kernel<<<1, 256, 0, stream>>>(top_i, counts, offsets, cursor);
  scatter_kernel<<<(NA + 255) / 256, 256, 0, stream>>>(top_i, top_w, offsets, cursor,
                                                       tokId, tokW);
  gateup_kernel<<<dim3(I_ / 128, TOKENS / 128, E_), 256, 0, stream>>>(
      xb, Wgt, Wut, bg, bu, offsets, tokId, hbuf);
  down_kernel<<<dim3(H_ / 128, TOKENS / 128, E_), 256, 0, stream>>>(
      hbuf, Wdt, bd, offsets, tokId, tokW, out);
}
